// DecoderBlock_71287867179088
// MI455X (gfx1250) — matrix operand order repaired and checked by reading, not compiled, not run on hardware
//
#include <hip/hip_runtime.h>

// ---------------------------------------------------------------------------
// MI455X (gfx1250, wave32) transformer decoder block.
// - All GEMMs on v_wmma_f32_16x16x32_bf16 (f32 accumulate), 64x64 wave tiles.
// - Global->LDS staging via CDNA5 async DMA (global_load_async_to_lds_b128,
//   ASYNCcnt) with double-buffered K-slabs.
// - Attention: flash-style single pass, per-wave online softmax.
// ---------------------------------------------------------------------------

typedef __attribute__((ext_vector_type(16))) __bf16 v16bf;
typedef __attribute__((ext_vector_type(8)))  float  v8f;

#define Bc   2
#define Sc   2048
#define Dc   2048
#define Hc   16
#define HDc  128
#define DFFc 8192
#define Mr   (Bc * Sc)          // 4096 rows of activations

// ----------------------------- helpers --------------------------------------

__device__ __forceinline__ v8f vzero8() {
  v8f z;
#pragma unroll
  for (int i = 0; i < 8; ++i) z[i] = 0.f;
  return z;
}

__device__ __forceinline__ v8f wmma_bf16(v16bf a, v16bf b, v8f c) {
  // (neg_a, A, neg_b, B, c_mod, C, reuse_a, reuse_b)
  return __builtin_amdgcn_wmma_f32_16x16x32_bf16(false, a, false, b,
                                                 (short)0, c, false, false);
}

// Per-lane async copy of 16 bytes global -> LDS (GV mode, tracked by ASYNCcnt).
__device__ __forceinline__ void async_b128(unsigned lds_off, const void* g) {
  asm volatile("global_load_async_to_lds_b128 %0, %1, off"
               :: "v"(lds_off), "v"(g) : "memory");
}
__device__ __forceinline__ unsigned lds_off32(const void* p) {
  return (unsigned)(unsigned long long)p;   // flat LDS addr keeps offset in [31:0]
}

// A-fragment (16x32 bf16, ISA 7.12.2): lane L<16 -> row L, K {0..7,16..23};
// lane L>=16 -> row L-16, K {8..15,24..31}. Two contiguous 16B loads.
__device__ __forceinline__ v16bf ld_afrag(const __bf16* base, int ld,
                                          int row0, int k0) {
  const int lane = threadIdx.x & 31;
  const int hl = lane >> 4, r = lane & 15;
  const __bf16* p = base + (size_t)(row0 + r) * ld + k0 + hl * 8;
  union { uint4 q[2]; v16bf v; } u;
  u.q[0] = *reinterpret_cast<const uint4*>(p);
  u.q[1] = *reinterpret_cast<const uint4*>(p + 16);
  return u.v;
}

// B-fragment (32x16 bf16) gathered from row-major W[N,K] (computes A @ W^T):
// lane L<16 -> W[n0+L][k0+{0..7,16..23}], lane>=16 -> W[n0+L-16][k0+{8..15,24..31}].
__device__ __forceinline__ v16bf ld_bfrag(const __bf16* base, int ld,
                                          int n0, int k0) {
  const int lane = threadIdx.x & 31;
  const int hl = lane >> 4, n = lane & 15;
  const __bf16* p = base + (size_t)(n0 + n) * ld + k0 + hl * 8;
  union { uint4 q[2]; v16bf v; } u;
  u.q[0] = *reinterpret_cast<const uint4*>(p);
  u.q[1] = *reinterpret_cast<const uint4*>(p + 16);
  return u.v;
}

// ----------------------------- small kernels --------------------------------

__global__ __launch_bounds__(256) void cast_f32_bf16(
    const float* __restrict__ in, __bf16* __restrict__ out, int n) {
  int i = blockIdx.x * 256 + threadIdx.x;
  const int stride = gridDim.x * 256;
  for (; i < n; i += stride) out[i] = (__bf16)in[i];
}

// One block per row: alpha * x / (sqrt(mean(x^2)) + eps), bf16 output.
__global__ __launch_bounds__(256) void rmsnorm_kernel(
    const float* __restrict__ x, const float* __restrict__ alpha,
    __bf16* __restrict__ out) {
  const int row = blockIdx.x;
  const int t = threadIdx.x;
  const float* xr = x + (size_t)row * Dc;
  float s = 0.f;
  for (int d = t; d < Dc; d += 256) { float v = xr[d]; s += v * v; }
#pragma unroll
  for (int mk = 1; mk < 32; mk <<= 1) s += __shfl_xor(s, mk, 32);
  __shared__ float red[8];
  if ((t & 31) == 0) red[t >> 5] = s;
  __syncthreads();
  float tot = 0.f;
#pragma unroll
  for (int i = 0; i < 8; ++i) tot += red[i];
  const float inv = 1.f / (sqrtf(tot * (1.f / Dc)) + 1e-8f);
  __bf16* orow = out + (size_t)row * Dc;
  for (int d = t; d < Dc; d += 256) orow[d] = (__bf16)(alpha[d] * xr[d] * inv);
}

// ----------------------------- tiled GEMM ----------------------------------
// C[M,N] = A[M,K](bf16) @ W[N,K](bf16)^T.
// Block tile 256x128, 8 waves as 4(m) x 2(n), wave tile 64x64 (4x4 WMMA).
// K staged 32-wide, double-buffered in LDS (stride 40 halfs: 16B-aligned,
// conflict-free ds_load_b128), filled by async global->LDS DMA.
// Per slab, per thread: 4 async ops (A) + 2 (W) = 6 -> steady-state
// s_wait_asynccnt 6 retires the previous slab (async loads complete in order).
// EPI 0: bf16 store.  EPI 1: +bias, SiLU, bf16 store.  EPI 2: +bias(+res) f32.

template <int EPI>
__global__ __launch_bounds__(256) void gemm_xwt(
    const __bf16* __restrict__ A, const __bf16* __restrict__ W,
    int M, int N, int K,
    const float* __restrict__ bias, const float* __restrict__ res,
    void* __restrict__ outv) {
  __shared__ __align__(16) __bf16 As[2][256 * 40];
  __shared__ __align__(16) __bf16 Ws[2][128 * 40];
  const int t = threadIdx.x;
  const int lane = t & 31, wid = t >> 5;
  const int wm = wid >> 1, wn = wid & 1;        // 4 x 2 wave grid
  const int row0 = blockIdx.y * 256, col0 = blockIdx.x * 128;

  v8f acc[4][4];
#pragma unroll
  for (int mt = 0; mt < 4; ++mt)
#pragma unroll
    for (int nt = 0; nt < 4; ++nt) acc[mt][nt] = vzero8();

  // Staging map: A -> one 256-row slab row per thread (32 halfs = 4 x 16B);
  //              W -> 128 rows, 2 threads/row (16 halfs = 2 x 16B each).
  const __bf16* gA = A + (size_t)(row0 + t) * K;
  const int rw = t >> 1, sw = (t & 1) * 16;
  const __bf16* gW = W + (size_t)(col0 + rw) * K + sw;
  const unsigned lA = lds_off32(&As[0][t * 40]);
  const unsigned lW = lds_off32(&Ws[0][rw * 40 + sw]);
  const unsigned szA = 256 * 40 * 2, szW = 128 * 40 * 2;

  // Prologue: stage slab 0 into buffer 0.
#pragma unroll
  for (int u = 0; u < 4; ++u) async_b128(lA + u * 16, gA + u * 8);
#pragma unroll
  for (int u = 0; u < 2; ++u) async_b128(lW + u * 16, gW + u * 8);

  for (int k0 = 0; k0 < K; k0 += 32) {
    const int buf = (k0 >> 5) & 1;
    if (k0 + 32 < K) {
      if (k0 + 64 < K) {   // global_prefetch_b8 one slab further out
        __builtin_prefetch(gA + k0 + 64, 0, 1);
        __builtin_prefetch(gW + k0 + 64, 0, 1);
      }
      const unsigned bo = (unsigned)(buf ^ 1);
#pragma unroll
      for (int u = 0; u < 4; ++u)
        async_b128(lA + bo * szA + u * 16, gA + k0 + 32 + u * 8);
#pragma unroll
      for (int u = 0; u < 2; ++u)
        async_b128(lW + bo * szW + u * 16, gW + k0 + 32 + u * 8);
      asm volatile("s_wait_asynccnt 0x6" ::: "memory");   // slab k0 resident
    } else {
      asm volatile("s_wait_asynccnt 0x0" ::: "memory");
    }
    __syncthreads();

    const __bf16* Ab = As[buf];
    const __bf16* Wb = Ws[buf];
    v16bf af[4], bf[4];
#pragma unroll
    for (int mt = 0; mt < 4; ++mt) af[mt] = ld_afrag(Ab, 40, wm * 64 + mt * 16, 0);
#pragma unroll
    for (int nt = 0; nt < 4; ++nt) bf[nt] = ld_bfrag(Wb, 40, wn * 64 + nt * 16, 0);
#pragma unroll
    for (int mt = 0; mt < 4; ++mt)
#pragma unroll
      for (int nt = 0; nt < 4; ++nt)
        acc[mt][nt] = wmma_bf16(af[mt], bf[nt], acc[mt][nt]);
    __syncthreads();   // all waves done with buf before it is restaged
  }

  // Epilogue. C layout: lane L, vgpr r -> row = r + 8*(L>>4), col = L&15.
  const int lh = lane >> 4, cl = lane & 15;
#pragma unroll
  for (int nt = 0; nt < 4; ++nt) {
    const int gc = col0 + wn * 64 + nt * 16 + cl;
    float bv = 0.f;
    if (EPI != 0) bv = bias ? bias[gc] : 0.f;
#pragma unroll
    for (int mt = 0; mt < 4; ++mt) {
      const int gr = row0 + wm * 64 + mt * 16 + 8 * lh;
#pragma unroll
      for (int rr = 0; rr < 8; ++rr) {
        float v = acc[mt][nt][rr] + bv;
        const size_t idx = (size_t)(gr + rr) * N + gc;
        if (EPI == 1) v = v / (1.f + __expf(-v));     // SiLU
        if (EPI == 2) {
          ((float*)outv)[idx] = res[idx] + v;         // residual, f32 out
        } else {
          ((__bf16*)outv)[idx] = (__bf16)v;           // bf16 out
        }
      }
    }
  }
}

// --------------------------- flash attention --------------------------------
// Grid: (S/128, B*H). Block = 256 threads = 8 waves; wave w owns 16 query
// rows, so online-softmax row stats live inside one 16-lane half.
// Per iteration: 64-key block; K rows async-DMA'd into LDS (stride 136,
// conflict-free), V staged transposed (Vt[d][k], stride 80) so P@V also uses
// the contiguous B-fragment path. P goes to per-wave LDS; only s_wait_dscnt
// is needed before re-reading it (same-wave RAW).

__global__ __launch_bounds__(256) void flash_attn_kernel(
    const __bf16* __restrict__ qkv, __bf16* __restrict__ attn) {
  __shared__ __align__(16) __bf16 Ks[64 * 136];    // 64 keys x 128 HD
  __shared__ __align__(16) __bf16 Vt[128 * 80];    // 128 HD x 64 keys
  __shared__ __align__(16) __bf16 Ps[128 * 80];    // 128 q rows x 64 keys

  const int t = threadIdx.x, lane = t & 31, w = t >> 5;
  const int lh = lane >> 4, cl = lane & 15;
  const int qb = blockIdx.x;
  const int b = blockIdx.y >> 4, h = blockIdx.y & 15;
  const int ldq = 3 * Dc;                           // qkv row stride = 6144
  const int qrow0 = b * Sc + qb * 128 + w * 16;

  v16bf qf[4];
#pragma unroll
  for (int kk = 0; kk < 4; ++kk)
    qf[kk] = ld_afrag(qkv, ldq, qrow0, h * HDc + kk * 32);

  v8f o[8];
#pragma unroll
  for (int dt = 0; dt < 8; ++dt) o[dt] = vzero8();
  float m_i[8], l_i[8];
#pragma unroll
  for (int rr = 0; rr < 8; ++rr) { m_i[rr] = -3.0e38f; l_i[rr] = 0.f; }

  const float scale = 0.08838834764831845f;         // 1/sqrt(128)
  const int nkb = 2 * qb + 2;                       // causal: keys <= qb*128+127
  const int kr = t >> 2, seg = (t & 3) * 32;
  const unsigned lK = lds_off32(&Ks[kr * 136 + seg]);

  for (int kb = 0; kb < nkb; ++kb) {
    // ---- stage K block (async DMA) and transposed V block into LDS ----
    {
      const __bf16* krow =
          qkv + (size_t)(b * Sc + kb * 64 + kr) * ldq + Dc + h * HDc + seg;
      const __bf16* vrow =
          qkv + (size_t)(b * Sc + kb * 64 + kr) * ldq + 2 * Dc + h * HDc + seg;
#pragma unroll
      for (int u = 0; u < 4; ++u) {
        async_b128(lK + u * 16, krow + u * 8);
        union { uint4 q; __bf16 e[8]; } uu;
        uu.q = *reinterpret_cast<const uint4*>(vrow + u * 8);
#pragma unroll
        for (int i = 0; i < 8; ++i) Vt[(seg + u * 8 + i) * 80 + kr] = uu.e[i];
      }
    }
    asm volatile("s_wait_asynccnt 0x0" ::: "memory");
    __syncthreads();

    // ---- scores: 16 q rows x 64 keys, K-dim = HD = 128 ----
    v8f sc[4];
#pragma unroll
    for (int nt = 0; nt < 4; ++nt) sc[nt] = vzero8();
#pragma unroll
    for (int kk = 0; kk < 4; ++kk)
#pragma unroll
      for (int nt = 0; nt < 4; ++nt)
        sc[nt] = wmma_bf16(qf[kk], ld_bfrag(Ks, 136, nt * 16, kk * 32), sc[nt]);

    // ---- causal mask + online softmax (stats per 16-lane half) ----
    float corr[8];
#pragma unroll
    for (int rr = 0; rr < 8; ++rr) {
      const int gi = qb * 128 + w * 16 + 8 * lh + rr;
      float rm = -3.0e38f;
#pragma unroll
      for (int nt = 0; nt < 4; ++nt) {
        const int gj = kb * 64 + nt * 16 + cl;
        float v = (gj <= gi) ? sc[nt][rr] * scale : -3.0e38f;
        sc[nt][rr] = v;
        rm = fmaxf(rm, v);
      }
#pragma unroll
      for (int mk = 1; mk < 16; mk <<= 1) rm = fmaxf(rm, __shfl_xor(rm, mk, 32));
      const float mn = fmaxf(m_i[rr], rm);
      corr[rr] = __expf(m_i[rr] - mn);
      float s = 0.f;
#pragma unroll
      for (int nt = 0; nt < 4; ++nt) {
        const float p = __expf(sc[nt][rr] - mn);   // masked -> exp(-inf) = 0
        sc[nt][rr] = p;
        s += p;
      }
#pragma unroll
      for (int mk = 1; mk < 16; mk <<= 1) s += __shfl_xor(s, mk, 32);
      l_i[rr] = l_i[rr] * corr[rr] + s;
      m_i[rr] = mn;
    }
#pragma unroll
    for (int dt = 0; dt < 8; ++dt)
#pragma unroll
      for (int rr = 0; rr < 8; ++rr) o[dt][rr] *= corr[rr];

    // ---- P -> per-wave LDS (bf16), same-wave RAW needs only dscnt wait ----
#pragma unroll
    for (int nt = 0; nt < 4; ++nt)
#pragma unroll
      for (int rr = 0; rr < 8; ++rr)
        Ps[(w * 16 + 8 * lh + rr) * 80 + nt * 16 + cl] = (__bf16)sc[nt][rr];
    asm volatile("s_wait_dscnt 0x0" ::: "memory");

    // ---- o += P @ V (K-dim = 64 keys) ----
#pragma unroll
    for (int kk = 0; kk < 2; ++kk) {
      const v16bf pa = ld_afrag(Ps, 80, w * 16, kk * 32);
#pragma unroll
      for (int dt = 0; dt < 8; ++dt)
        o[dt] = wmma_bf16(pa, ld_bfrag(Vt, 80, dt * 16, kk * 32), o[dt]);
    }
    __syncthreads();   // Ks/Vt reused next iteration
  }

  // ---- normalize and write [B,S,H*HD] bf16 ----
#pragma unroll
  for (int rr = 0; rr < 8; ++rr) {
    const float inv = 1.f / l_i[rr];
#pragma unroll
    for (int dt = 0; dt < 8; ++dt)
      attn[(size_t)(qrow0 + 8 * lh + rr) * Dc + h * HDc + dt * 16 + cl] =
          (__bf16)(o[dt][rr] * inv);
  }
}

// ------------------------------- launcher -----------------------------------

extern "C" void kernel_launch(void* const* d_in, const int* in_sizes, int n_in,
                              void* d_out, int out_size, void* d_ws,
                              size_t ws_size, hipStream_t stream) {
  (void)in_sizes; (void)n_in; (void)out_size; (void)ws_size;
  const float* x      = (const float*)d_in[0];
  // d_in[1] = freqs_complex_form (unused), d_in[2] = tgt_mask (causal, analytic)
  const float* wq     = (const float*)d_in[3];
  const float* wk     = (const float*)d_in[4];
  const float* wv     = (const float*)d_in[5];
  const float* wo     = (const float*)d_in[6];
  const float* alpha1 = (const float*)d_in[7];
  const float* alpha3 = (const float*)d_in[8];
  const float* fc1w   = (const float*)d_in[9];
  const float* fc1b   = (const float*)d_in[10];
  const float* fc2w   = (const float*)d_in[11];
  const float* fc2b   = (const float*)d_in[12];
  float* out = (float*)d_out;

  char* ws = (char*)d_ws;
  size_t off = 0;
  auto carve = [&](size_t bytes) -> void* {
    void* p = ws + off;
    off += (bytes + 255) & ~(size_t)255;
    return p;
  };
  const size_t DD = (size_t)Dc * Dc;
  __bf16* wqkvB = (__bf16*)carve(3 * DD * 2);            // [6144,2048] bf16
  __bf16* woB   = (__bf16*)carve(DD * 2);                // [2048,2048]
  __bf16* fc1B  = (__bf16*)carve((size_t)DFFc * Dc * 2); // [8192,2048]
  __bf16* fc2B  = (__bf16*)carve((size_t)Dc * DFFc * 2); // [2048,8192]
  __bf16* normA = (__bf16*)carve((size_t)Mr * Dc * 2);
  __bf16* qkv   = (__bf16*)carve((size_t)Mr * 3 * Dc * 2);
  __bf16* attn  = (__bf16*)carve((size_t)Mr * Dc * 2);
  float*  x1    = (float*)carve((size_t)Mr * Dc * 4);
  __bf16* normB = (__bf16*)carve((size_t)Mr * Dc * 2);
  __bf16* hbuf  = (__bf16*)carve((size_t)Mr * DFFc * 2);

  // 1) weights -> bf16 (wq/wk/wv fused into one [3D, D] matrix)
  cast_f32_bf16<<<1024, 256, 0, stream>>>(wq, wqkvB, (int)DD);
  cast_f32_bf16<<<1024, 256, 0, stream>>>(wk, wqkvB + DD, (int)DD);
  cast_f32_bf16<<<1024, 256, 0, stream>>>(wv, wqkvB + 2 * DD, (int)DD);
  cast_f32_bf16<<<1024, 256, 0, stream>>>(wo, woB, (int)DD);
  cast_f32_bf16<<<2048, 256, 0, stream>>>(fc1w, fc1B, DFFc * Dc);
  cast_f32_bf16<<<2048, 256, 0, stream>>>(fc2w, fc2B, Dc * DFFc);

  // 2) norm1 = rmsnorm(x, alpha1)
  rmsnorm_kernel<<<Mr, 256, 0, stream>>>(x, alpha1, normA);

  // 3) qkv = norm1 @ [wq;wk;wv]^T            [4096 x 6144] bf16
  gemm_xwt<0><<<dim3(48, 16), 256, 0, stream>>>(normA, wqkvB, Mr, 3 * Dc, Dc,
                                                nullptr, nullptr, qkv);

  // 4) flash attention -> [B,S,D] bf16
  flash_attn_kernel<<<dim3(Sc / 128, Bc * Hc), 256, 0, stream>>>(qkv, attn);

  // 5) x1 = x + attn @ wo^T                  [4096 x 2048] f32
  gemm_xwt<2><<<dim3(16, 16), 256, 0, stream>>>(attn, woB, Mr, Dc, Dc,
                                                nullptr, x, x1);

  // 6) norm3 = rmsnorm(x1, alpha3)
  rmsnorm_kernel<<<Mr, 256, 0, stream>>>(x1, alpha3, normB);

  // 7) h = silu(norm3 @ fc1^T + fc1_b)       [4096 x 8192] bf16
  gemm_xwt<1><<<dim3(64, 16), 256, 0, stream>>>(normB, fc1B, Mr, DFFc, Dc,
                                                fc1b, nullptr, hbuf);

  // 8) out = x1 + h @ fc2^T + fc2_b          [4096 x 2048] f32
  gemm_xwt<2><<<dim3(16, 16), 256, 0, stream>>>(hbuf, fc2B, Mr, Dc, DFFc,
                                                fc2b, x1, out);
}
